// CosineMatrix_49168785605045
// MI455X (gfx1250) — matrix operand order repaired and checked by reading, not compiled, not run on hardware
//
#include <hip/hip_runtime.h>
#include <hip/hip_bf16.h>
#include <math.h>

typedef __attribute__((ext_vector_type(16))) __bf16 v16bf;
typedef __attribute__((ext_vector_type(8)))  __bf16 v8bf;
typedef __attribute__((ext_vector_type(4)))  __bf16 v4bf;
typedef __attribute__((ext_vector_type(8)))  float  v8f;
typedef __attribute__((ext_vector_type(4)))  unsigned int v4u;
typedef __attribute__((ext_vector_type(8)))  int v8i;
typedef __attribute__((ext_vector_type(4)))  int v4i;

#define BATCH 8
#define NROW  2048
#define DDIM  512
#define BMR   256              // X rows per block tile
#define BNR   128              // Y rows per block tile
#define BK    32
#define RP    (2 * DDIM)       // hi/lo-interleaved row pitch (elements)
#define CEPS  1e-8f

// ---------------------------------------------------------------------------
// Kernel 1: row norms.  One wave32 per 512-float row.
// ---------------------------------------------------------------------------
__global__ __launch_bounds__(256) void cosmat_norms_kernel(
    const float* __restrict__ x, const float* __restrict__ y,
    float* __restrict__ xn, float* __restrict__ yn) {
  const int ROWS  = BATCH * NROW;
  const int gwave = (int)((blockIdx.x * 256u + threadIdx.x) >> 5);
  const int lane  = threadIdx.x & 31;
  if (gwave >= 2 * ROWS) return;

  const float* src;
  float* dst;
  int row;
  if (gwave < ROWS) { row = gwave;        src = x + (size_t)row * DDIM; dst = xn; }
  else              { row = gwave - ROWS; src = y + (size_t)row * DDIM; dst = yn; }

  const float4* s4 = (const float4*)src;
  float s = 0.0f;
#pragma unroll
  for (int k = 0; k < DDIM / 128; ++k) {
    float4 v = s4[lane + 32 * k];
    s += v.x * v.x + v.y * v.y + v.z * v.z + v.w * v.w;
  }
#pragma unroll
  for (int off = 16; off > 0; off >>= 1) s += __shfl_xor(s, off, 32);
  if (lane == 0) dst[row] = sqrtf(s);
}

// ---------------------------------------------------------------------------
// Kernel 2: one-pass split-bf16 conversion.  fp32 [row][512] ->
// bf16 [row][2][512] with plane 0 = hi = bf16(f), plane 1 = lo = bf16(f - hi).
// ---------------------------------------------------------------------------
__global__ __launch_bounds__(256) void cosmat_cvt_kernel(
    const float* __restrict__ x, const float* __restrict__ y,
    __bf16* __restrict__ Xhl, __bf16* __restrict__ Yhl) {
  const size_t tg  = (size_t)blockIdx.x * 256 + threadIdx.x;  // 4 elems / thread
  const float*  src = blockIdx.y ? y : x;
  __bf16*       dst = blockIdx.y ? Yhl : Xhl;
  const size_t row = tg >> 7;            // 128 threads per 512-row
  const int    kq  = ((int)tg & 127) * 4;

  float4 v = *(const float4*)&src[row * DDIM + kq];
  float f[4] = {v.x, v.y, v.z, v.w};
  v4bf h, l;
#pragma unroll
  for (int q = 0; q < 4; ++q) {
    h[q] = (__bf16)f[q];
    l[q] = (__bf16)(f[q] - (float)h[q]);
  }
  *(v4bf*)&dst[(row * 2 + 0) * DDIM + kq] = h;
  *(v4bf*)&dst[(row * 2 + 1) * DDIM + kq] = l;
}

// ---------------------------------------------------------------------------
// TDM: load a 3-D tile  (dim0 = BK elems, dim1 = 2 hi/lo planes, dim2 = rows)
// from the interleaved bf16 panel into LDS.  D# packing per CDNA5 ISA ch.8.
// ---------------------------------------------------------------------------
__device__ __forceinline__ unsigned lds_addr_of(const void* p) {
  return (unsigned)(uintptr_t)p;        // low 32 bits of flat addr = LDS offset
}

__device__ __forceinline__ void tdm_load_tile(const __bf16* gtile,
                                              unsigned lds_byte, int rows) {
  unsigned long long ga = (unsigned long long)(uintptr_t)gtile;
  v4u g0 = {1u,                                   // count = 1 valid descriptor
            lds_byte,                             // LDS byte address
            (unsigned)ga,                         // global_addr[31:0]
            ((unsigned)(ga >> 32) & 0x01FFFFFFu)  // global_addr[56:32]
                | 0x80000000u};                   // type = 2 ("image")
  v8i g1;
  g1[0] = (int)(1u << 16);                               // data_size = 2 bytes
  g1[1] = (int)(((unsigned)DDIM & 0xFFFFu) << 16);       // tensor_dim0 lo16 = 512
  g1[2] = (int)(((unsigned)DDIM >> 16) | (2u << 16));    // dim0 hi | tensor_dim1 = 2
  g1[3] = (int)((unsigned)BK << 16);                     // dim1 hi | tile_dim0 = 32
  g1[4] = (int)(2u | ((unsigned)rows << 16));            // tile_dim1 = 2 | tile_dim2
  g1[5] = (int)DDIM;                                     // tensor_dim0_stride = 512
  g1[6] = (int)(((unsigned)RP & 0xFFFFu) << 16);         // dim1_stride lo16 = 1024
  g1[7] = (int)((unsigned)RP >> 16);                     // dim1_stride hi = 0
  v4i g2 = {NROW, 0, 0, 0};                              // tensor_dim2 = 2048
  v4i g3 = {0, 0, 0, 0};
#if defined(__clang_major__) && (__clang_major__ >= 23)
  v8i g4 = {0, 0, 0, 0, 0, 0, 0, 0};
  __builtin_amdgcn_tensor_load_to_lds(g0, g1, g2, g3, g4, 0);
#else
  __builtin_amdgcn_tensor_load_to_lds(g0, g1, g2, g3, 0);
#endif
}

// ---------------------------------------------------------------------------
// LDS operand fetch for 16-bit WMMA; A and B both use contiguous k = 16*hb + s.
// LDS tile layout: [row][hl][BK] bf16, row pitch 2*BK = 64 elements.
// ---------------------------------------------------------------------------
__device__ __forceinline__ v16bf load_a(const __bf16* s, int row0, int hl,
                                        int l16, int hb) {
  const __bf16* row = s + (row0 + l16) * (2 * BK) + hl * BK;
  return *(const v16bf*)(row + hb * 16);         // K {0..15} / {16..31}
}
__device__ __forceinline__ v16bf load_b(const __bf16* s, int row0, int hl,
                                        int l16, int hb) {
  return *(const v16bf*)(s + (row0 + l16) * (2 * BK) + hl * BK + hb * 16);
}

// ---------------------------------------------------------------------------
// Kernel 3: 256x128 output tile per block, 8 waves (4x2), 64x64 per wave.
// TDM double-buffered staging; split-bf16 WMMA: acc += Ah*Bh + Ah*Bl + Al*Bh.
// ---------------------------------------------------------------------------
__global__ __launch_bounds__(256) void cosmat_gemm_kernel(
    const __bf16* __restrict__ Xhl, const __bf16* __restrict__ Yhl,
    const float* __restrict__ xn, const float* __restrict__ yn,
    float* __restrict__ out) {
  __shared__ __bf16 sX[2][BMR * 2 * BK];   // 2 x 32 KB
  __shared__ __bf16 sY[2][BNR * 2 * BK];   // 2 x 16 KB
  __shared__ float  xn_s[BMR];
  __shared__ float  yn_s[BNR];

  const int tid = threadIdx.x;
  const int b = blockIdx.z, bm = blockIdx.x, bn = blockIdx.y;

  const size_t rowX = (size_t)b * NROW + (size_t)bm * BMR;
  const size_t rowY = (size_t)b * NROW + (size_t)bn * BNR;

  xn_s[tid] = xn[rowX + tid];
  if (tid < BNR) yn_s[tid] = yn[rowY + tid];

  const __bf16* gX = Xhl + rowX * RP;      // tile start, advance by kk elems
  const __bf16* gY = Yhl + rowY * RP;

  if (tid < 32) {                          // wave 0 drives the TDM
    tdm_load_tile(gX, lds_addr_of(&sX[0][0]), BMR);
    tdm_load_tile(gY, lds_addr_of(&sY[0][0]), BNR);
  }

  const int w    = tid >> 5;
  const int wm   = (w >> 1) * 64;          // 4 wave-rows over 256
  const int wn   = (w & 1) * 64;           // 2 wave-cols over 128
  const int lane = tid & 31;
  const int l16  = lane & 15;
  const int hb   = lane >> 4;

  v8f acc[4][4] = {};

  for (int it = 0; it < DDIM / BK; ++it) {
    if (tid < 32) __builtin_amdgcn_s_wait_tensorcnt((short)0);
    __syncthreads();                       // buffer it&1 visible; it^1 free
    if (tid < 32 && it + 1 < DDIM / BK) {
      const int nb = (it + 1) & 1;
      const int kk = (it + 1) * BK;
      tdm_load_tile(gX + kk, lds_addr_of(&sX[nb][0]), BMR);
      tdm_load_tile(gY + kk, lds_addr_of(&sY[nb][0]), BNR);
    }
    const __bf16* bx = sX[it & 1];
    const __bf16* by = sY[it & 1];

    v16bf ah[4], al[4];
#pragma unroll
    for (int i = 0; i < 4; ++i) {
      ah[i] = load_a(bx, wm + i * 16, 0, l16, hb);
      al[i] = load_a(bx, wm + i * 16, 1, l16, hb);
    }
#pragma unroll
    for (int j = 0; j < 4; ++j) {
      v16bf bh = load_b(by, wn + j * 16, 0, l16, hb);
      v16bf bl = load_b(by, wn + j * 16, 1, l16, hb);
#pragma unroll
      for (int i = 0; i < 4; ++i) {
        acc[i][j] = __builtin_amdgcn_wmma_f32_16x16x32_bf16(
            false, ah[i], false, bh, (short)0, acc[i][j], false, false);
        acc[i][j] = __builtin_amdgcn_wmma_f32_16x16x32_bf16(
            false, ah[i], false, bl, (short)0, acc[i][j], false, false);
        acc[i][j] = __builtin_amdgcn_wmma_f32_16x16x32_bf16(
            false, al[i], false, bh, (short)0, acc[i][j], false, false);
      }
    }
  }

  // Epilogue: C layout M = r + 8*hb, N = l16.  out = acc / max(xn*yn, eps).
#pragma unroll
  for (int i = 0; i < 4; ++i) {
    const int mt = wm + i * 16;
    float xv[8];
#pragma unroll
    for (int r = 0; r < 8; ++r) xv[r] = xn_s[mt + hb * 8 + r];
#pragma unroll
    for (int j = 0; j < 4; ++j) {
      const int nt   = wn + j * 16;
      const float yv = yn_s[nt + l16];
      const size_t col = (size_t)bn * BNR + nt + l16;
#pragma unroll
      for (int r = 0; r < 8; ++r) {
        const size_t m     = (size_t)bm * BMR + mt + hb * 8 + r;
        const float  denom = fmaxf(xv[r] * yv, CEPS);
        const size_t idx   = ((size_t)b * NROW + m) * (size_t)NROW + col;
        __builtin_nontemporal_store(acc[i][j][r] / denom, &out[idx]);
      }
    }
  }
}

// ---------------------------------------------------------------------------
extern "C" void kernel_launch(void* const* d_in, const int* in_sizes, int n_in,
                              void* d_out, int out_size, void* d_ws, size_t ws_size,
                              hipStream_t stream) {
  (void)in_sizes; (void)n_in; (void)out_size; (void)ws_size;
  const float* x = (const float*)d_in[0];
  const float* y = (const float*)d_in[1];
  float* out = (float*)d_out;

  // workspace: xn[16384] f32, yn[16384] f32 (128 KB), then bf16 hi/lo panels.
  float*  xn  = (float*)d_ws;
  float*  yn  = xn + BATCH * NROW;
  __bf16* Xhl = (__bf16*)((char*)d_ws + (1u << 17));
  __bf16* Yhl = Xhl + (size_t)BATCH * NROW * RP;     // +33.5 MB

  cosmat_norms_kernel<<<(2 * BATCH * NROW) / 8, 256, 0, stream>>>(x, y, xn, yn);

  dim3 cgrid(BATCH * NROW * DDIM / (256 * 4), 2);    // (8192, 2)
  cosmat_cvt_kernel<<<cgrid, 256, 0, stream>>>(x, y, Xhl, Yhl);

  dim3 grid(NROW / BMR, NROW / BNR, BATCH);          // (8, 16, 8)
  cosmat_gemm_kernel<<<grid, 256, 0, stream>>>(Xhl, Yhl, xn, yn, out);
}
